// TFCAM_54520314856359
// MI455X (gfx1250) — matrix operand order repaired and checked by reading, not compiled, not run on hardware
//
#include <hip/hip_runtime.h>
#include <hip/hip_bf16.h>
#include <math.h>

// ---------------------------------------------------------------------------
// TFCAM forward for MI455X (gfx1250, wave32, WMMA).
//   B=16 T=32 F=64 D=128 H=128 NH=8 DH=16 NL=2 NC=1; BT=512
// Roofline: influence tensor output = 268 MB -> ~11.5us at 23.3 TB/s HBM;
// all GEMMs (~1 GFLOP) are L2/LDS resident. GEMMs: v_wmma_f32_16x16x32_f16
// with async-to-LDS tile staging (ASYNCcnt). Influence: full-wave contiguous
// nontemporal float2 streaming stores. LSTM: persistent blocks, LDS state,
// cooperative f16 weight-panel staging.
// ---------------------------------------------------------------------------

typedef __attribute__((ext_vector_type(16))) _Float16 v16h;
typedef __attribute__((ext_vector_type(8)))  float    v8f;
typedef __attribute__((ext_vector_type(2)))  float    vf2;

__device__ __forceinline__ float sigf(float x) { return 1.0f / (1.0f + __expf(-x)); }

// A-matrix (16-bit, 16x32) per-lane element e -> K index (ISA 7.12.2)
__device__ __forceinline__ int a_kmap(int e, int lanegrp) {
    return ((e >> 3) << 4) + (lanegrp << 3) + (e & 7);
}
// B-matrix (16-bit, 32x16): same K order as A (e<8: 8*lanegrp+e; e>=8: 16+8*lanegrp+e-8)
__device__ __forceinline__ int b_kmap(int e, int lanegrp) {
    return ((e >> 3) << 4) + (lanegrp << 3) + (e & 7);
}

// Async DMA: 16B global -> LDS, tracked on ASYNCcnt (VGLOBAL op 98).
__device__ __forceinline__ void async_b128(unsigned lds_off, const void* gaddr) {
    asm volatile("global_load_async_to_lds_b128 %0, %1, off"
                 :: "v"(lds_off), "v"(gaddr) : "memory");
}
__device__ __forceinline__ void wait_async0() {
    asm volatile("s_wait_asynccnt 0" ::: "memory");
}
__device__ __forceinline__ unsigned lds_off32(const void* p) {
    return (unsigned)(uintptr_t)p;     // low 32 bits of flat addr == LDS offset
}

// ---------------------------------------------------------------------------
// f32 -> f16 weight conversion with zero row padding (pad indices >= n_src).
// ---------------------------------------------------------------------------
__global__ __launch_bounds__(256) void k_cvt(const float* __restrict__ src,
                                             _Float16* __restrict__ dst,
                                             int n_src, int n_total)
{
    int i = blockIdx.x * 256 + threadIdx.x;
    if (i < n_total) dst[i] = (i < n_src) ? (_Float16)src[i] : (_Float16)0.0f;
}

// ---------------------------------------------------------------------------
// Generic GEMM: out[M,N] = act(A[M,K](f32) @ W16[Npad,K](f16)^T + b1 + b2)
// block = 256 thr (8 waves); tile = 16(M) x 128(N); K mult of 32; W16 rows
// zero-padded to a multiple of 128 so staging is guard-free.
// act: 0 none, 1 relu, 2 tanh
// ---------------------------------------------------------------------------
__global__ __launch_bounds__(256) void k_gemm(
    const float* __restrict__ A, const _Float16* __restrict__ W16,
    const float* __restrict__ b1, const float* __restrict__ b2,
    float* __restrict__ out, int M, int N, int K, int act)
{
    __shared__ float    sA[16 * 32];      // async-staged f32 activations
    __shared__ _Float16 sW[128 * 32];     // async-staged f16 weights
    const int tid  = threadIdx.x;
    const int wave = tid >> 5, lane = tid & 31;
    const int mrow = lane & 15, lg = lane >> 4;
    const int m0   = blockIdx.x * 16;
    const int nblk = blockIdx.y * 128;
    const int n0   = nblk + wave * 16;
    const unsigned sA_b = lds_off32(&sA[0]);
    const unsigned sW_b = lds_off32(&sW[0]);

    v8f acc = {};
    for (int k0 = 0; k0 < K; k0 += 32) {
        if (tid < 128) {                                   // A: 128 x 16B chunks
            int m = tid >> 3, k4 = (tid & 7) * 4;
            async_b128(sA_b + (unsigned)(m * 32 + k4) * 4,
                       A + (size_t)(m0 + m) * K + k0 + k4);
        }
        for (int c = tid; c < 512; c += 256) {             // W: 512 x 16B chunks
            int n = c >> 2, k8 = (c & 3) * 8;
            async_b128(sW_b + (unsigned)(n * 32 + k8) * 2,
                       W16 + (size_t)(nblk + n) * K + k0 + k8);
        }
        wait_async0();
        __syncthreads();
        v16h af, bf;
#pragma unroll
        for (int e = 0; e < 16; ++e) {
            af[e] = (_Float16)sA[mrow * 32 + a_kmap(e, lg)];
            bf[e] = sW[(wave * 16 + mrow) * 32 + b_kmap(e, lg)];
        }
        acc = __builtin_amdgcn_wmma_f32_16x16x32_f16(false, af, false, bf,
                                                     (short)0, acc, false, false);
        __syncthreads();
    }
    if (n0 < N) {
        int ncol = n0 + mrow;
        float bias = (b1 ? b1[ncol] : 0.0f) + (b2 ? b2[ncol] : 0.0f);
#pragma unroll
        for (int r = 0; r < 8; ++r) {
            int m = m0 + r + (lg << 3);
            float v = acc[r] + bias;
            if (act == 1) v = fmaxf(v, 0.0f);
            else if (act == 2) v = tanhf(v);
            out[(size_t)m * N + ncol] = v;
        }
    }
}

// E += positional encoding (E is (B*T, D), t = row % T)
__global__ __launch_bounds__(256) void k_add_pe(float* __restrict__ E)
{
    int idx = blockIdx.x * 256 + threadIdx.x;          // < 512*128
    int row = idx >> 7, n = idx & 127;
    int t = row & 31;
    int pair = n >> 1;
    float w = __expf(-(float)(2 * pair) * (9.210340371976184f / 128.0f));
    float a = (float)t * w;
    E[idx] += (n & 1) ? __cosf(a) : __sinf(a);
}

// ---------------------------------------------------------------------------
// 4 persistent LSTM directions (blockIdx.x = dir):
//   0: la_f -> out_a[:,:,0:128]  1: la_r -> out_a[:,:,128:256]
//   2: lb_f -> out_b[:,:,0:128]  3: lb_r -> out_b[:,:,128:256]
// gates = xproj[t] + h @ whh16^T. Per k-step a 256x32 f16 weight panel is
// async-staged to LDS (coalesced b128); each wave keeps 2 WMMA accumulators.
// ---------------------------------------------------------------------------
__global__ __launch_bounds__(256) void k_lstm(
    const float* __restrict__ xp0, const float* __restrict__ xp1,
    const float* __restrict__ xp2, const float* __restrict__ xp3,
    const _Float16* __restrict__ w0, const _Float16* __restrict__ w1,
    const _Float16* __restrict__ w2, const _Float16* __restrict__ w3,
    float* __restrict__ out_a, float* __restrict__ out_b)
{
    const int dir = blockIdx.x;
    const float* xp = (dir == 0) ? xp0 : (dir == 1) ? xp1 : (dir == 2) ? xp2 : xp3;
    const _Float16* whh16 = (dir == 0) ? w0 : (dir == 1) ? w1 : (dir == 2) ? w2 : w3;
    float* out = (dir < 2) ? out_a : out_b;
    const int co  = (dir & 1) ? 128 : 0;
    const int rev = dir & 1;

    __shared__ _Float16 sh[16 * 128];     // h state (f16, WMMA A)        4KB
    __shared__ float    sc[16 * 128];     // c state                      8KB
    __shared__ float    sg[16 * 512];     // gates                       32KB
    __shared__ _Float16 swt[256 * 32];    // weight panel (async)        16KB
    const int tid = threadIdx.x;
    const int wave = tid >> 5, lane = tid & 31;
    const int mrow = lane & 15, lg = lane >> 4;
    const unsigned swt_b = lds_off32(&swt[0]);

    for (int i = tid; i < 2048; i += 256) { sh[i] = (_Float16)0.0f; sc[i] = 0.0f; }
    __syncthreads();

    for (int s = 0; s < 32; ++s) {
        int t = rev ? (31 - s) : s;
        for (int g = 0; g < 2; ++g) {                     // gate halves (256 cols)
            v8f acc0 = {}, acc1 = {};
            for (int k0 = 0; k0 < 128; k0 += 32) {
                for (int c = tid; c < 1024; c += 256) {   // 256x32 halves, 16B chunks
                    int n = c >> 2, k8 = (c & 3) * 8;
                    async_b128(swt_b + (unsigned)(n * 32 + k8) * 2,
                               whh16 + (size_t)(g * 256 + n) * 128 + k0 + k8);
                }
                wait_async0();
                __syncthreads();
                v16h af, bf0, bf1;
#pragma unroll
                for (int e = 0; e < 16; ++e) {
                    af[e] = sh[mrow * 128 + k0 + a_kmap(e, lg)];
                    int kb = b_kmap(e, lg);
                    bf0[e] = swt[(wave * 32 + mrow) * 32 + kb];
                    bf1[e] = swt[(wave * 32 + 16 + mrow) * 32 + kb];
                }
                acc0 = __builtin_amdgcn_wmma_f32_16x16x32_f16(false, af, false, bf0,
                                                              (short)0, acc0, false, false);
                acc1 = __builtin_amdgcn_wmma_f32_16x16x32_f16(false, af, false, bf1,
                                                              (short)0, acc1, false, false);
                __syncthreads();
            }
            int ncol = g * 256 + wave * 32 + mrow;
#pragma unroll
            for (int r = 0; r < 8; ++r) {
                int m = r + (lg << 3);                    // batch index
                const float* xr = xp + (m * 32 + t) * 512;
                sg[m * 512 + ncol]      = acc0[r] + xr[ncol];
                sg[m * 512 + ncol + 16] = acc1[r] + xr[ncol + 16];
            }
        }
        __syncthreads();
        // state update: gate order i,f,g,o in chunks of 128
        for (int e = tid; e < 2048; e += 256) {
            int m = e >> 7, hh = e & 127;
            float ig = sg[m * 512 + hh];
            float fg = sg[m * 512 + 128 + hh];
            float gg = sg[m * 512 + 256 + hh];
            float og = sg[m * 512 + 384 + hh];
            float c = sigf(fg) * sc[e] + sigf(ig) * tanhf(gg);
            float h = sigf(og) * tanhf(c);
            sc[e] = c;
            sh[e] = (_Float16)h;
            out[(m * 32 + t) * 256 + co + hh] = h;
        }
        __syncthreads();
    }
}

// alpha = softmax_T(out_a @ fa_w^T + fa_b); one block per batch
__global__ __launch_bounds__(256) void k_alpha(
    const float* __restrict__ outa, const float* __restrict__ faw,
    const float* __restrict__ fab, float* __restrict__ alpha)
{
    const int b = blockIdx.x, tid = threadIdx.x;
    const int t = tid >> 3, sub = tid & 7;
    __shared__ float ss[32];
    __shared__ float red[2];
    float p = 0.0f;
    for (int k = sub; k < 256; k += 8) p += outa[(b * 32 + t) * 256 + k] * faw[k];
    p += __shfl_down(p, 4, 8);
    p += __shfl_down(p, 2, 8);
    p += __shfl_down(p, 1, 8);
    if (sub == 0) ss[t] = p + fab[0];
    __syncthreads();
    if (tid == 0) {
        float mx = -1e30f;
        for (int i = 0; i < 32; ++i) mx = fmaxf(mx, ss[i]);
        float sm = 0.0f;
        for (int i = 0; i < 32; ++i) sm += __expf(ss[i] - mx);
        red[0] = mx; red[1] = sm;
    }
    __syncthreads();
    if (tid < 32) alpha[b * 32 + tid] = __expf(ss[tid] - red[0]) / red[1];
}

// C[b,t,f] = alpha[b,t] * beta[b,t,f] * x[b,t,f]
__global__ __launch_bounds__(256) void k_C(
    const float* __restrict__ x, const float* __restrict__ alpha,
    const float* __restrict__ beta, float* __restrict__ C)
{
    int idx = blockIdx.x * 256 + threadIdx.x;          // < 32768
    int bt = idx >> 6;
    C[idx] = alpha[bt] * beta[idx] * x[idx];
}

// ---------------------------------------------------------------------------
// Attention core: one block per batch, loop over 8 heads. T=32, DH=16.
// Accumulates head/layer-averaged attention into Aout (write L0, add L1).
// ---------------------------------------------------------------------------
__global__ __launch_bounds__(256) void k_attn(
    const float* __restrict__ qkv, float* __restrict__ o,
    float* __restrict__ Aout, int layer)
{
    const int b = blockIdx.x, tid = threadIdx.x;
    __shared__ float sq[512], sk[512], sv[512], ss[1024], sacc[1024];
    for (int i = tid; i < 1024; i += 256) sacc[i] = 0.0f;
    for (int h = 0; h < 8; ++h) {
        __syncthreads();
        for (int i = tid; i < 512; i += 256) {
            int t = i >> 4, d = i & 15;
            int base = (b * 32 + t) * 384 + h * 16 + d;
            sq[i] = qkv[base];
            sk[i] = qkv[base + 128];
            sv[i] = qkv[base + 256];
        }
        __syncthreads();
        for (int i = tid; i < 1024; i += 256) {
            int t = i >> 5, u = i & 31;
            float s = 0.0f;
            for (int d = 0; d < 16; ++d) s += sq[t * 16 + d] * sk[u * 16 + d];
            ss[i] = s * 0.25f;                          // 1/sqrt(DH)
        }
        __syncthreads();
        if (tid < 32) {
            float mx = -1e30f;
            for (int u = 0; u < 32; ++u) mx = fmaxf(mx, ss[tid * 32 + u]);
            float sm = 0.0f;
            for (int u = 0; u < 32; ++u) { float e = __expf(ss[tid * 32 + u] - mx); ss[tid * 32 + u] = e; sm += e; }
            float inv = 1.0f / sm;
            for (int u = 0; u < 32; ++u) ss[tid * 32 + u] *= inv;
        }
        __syncthreads();
        for (int i = tid; i < 1024; i += 256) sacc[i] += ss[i] * (1.0f / 16.0f); // 1/(NH*NL)
        for (int i = tid; i < 512; i += 256) {
            int t = i >> 4, d = i & 15;
            float s = 0.0f;
            for (int u = 0; u < 32; ++u) s += ss[t * 32 + u] * sv[u * 16 + d];
            o[(b * 32 + t) * 128 + h * 16 + d] = s;
        }
    }
    __syncthreads();
    for (int i = tid; i < 1024; i += 256) {
        if (layer == 0) Aout[b * 1024 + i] = sacc[i];
        else            Aout[b * 1024 + i] += sacc[i];
    }
}

// out = layernorm(x + r) * g + bta ; one block (128 thr) per row
__global__ __launch_bounds__(128) void k_addln(
    const float* __restrict__ x, const float* __restrict__ r,
    const float* __restrict__ g, const float* __restrict__ bta,
    float* __restrict__ out)
{
    const int row = blockIdx.x, tid = threadIdx.x;
    __shared__ float s1[128];
    float v = x[row * 128 + tid] + r[row * 128 + tid];
    s1[tid] = v;
    __syncthreads();
    for (int st = 64; st > 0; st >>= 1) { if (tid < st) s1[tid] += s1[tid + st]; __syncthreads(); }
    float mu = s1[0] * (1.0f / 128.0f);
    __syncthreads();
    float d = v - mu;
    s1[tid] = d * d;
    __syncthreads();
    for (int st = 64; st > 0; st >>= 1) { if (tid < st) s1[tid] += s1[tid + st]; __syncthreads(); }
    float var = s1[0] * (1.0f / 128.0f);
    out[row * 128 + tid] = d * rsqrtf(var + 1e-5f) * g[tid] + bta[tid];
}

// context[b,d] = sum_t xc[b,t,d] * alpha[b,t] ; one block (128 thr) per batch
__global__ __launch_bounds__(128) void k_context(
    const float* __restrict__ xc, const float* __restrict__ alpha,
    float* __restrict__ ctx)
{
    const int b = blockIdx.x, d = threadIdx.x;
    float s = 0.0f;
    for (int t = 0; t < 32; ++t) s += xc[(b * 32 + t) * 128 + d] * alpha[b * 32 + t];
    ctx[b * 128 + d] = s;
}

// logits[b] = h[b,:] @ c2_w^T + c2_b  (NC=1)
__global__ __launch_bounds__(32) void k_logits(
    const float* __restrict__ h, const float* __restrict__ c2w,
    const float* __restrict__ c2b, float* __restrict__ logits)
{
    int b = threadIdx.x;
    if (b < 16) {
        float s = c2b[0];
        for (int j = 0; j < 64; ++j) s += h[b * 64 + j] * c2w[j];
        logits[b] = s;
    }
}

// ---------------------------------------------------------------------------
// influence[b,t,i,u,j] = C[b,t,i] * A[b,t,u] * C[b,u,j]
// 268 MB streaming output. One block per (b,t,u); each wave writes full
// contiguous 256B rows with nontemporal float2 stores (bypass L2 residency).
// ---------------------------------------------------------------------------
__global__ __launch_bounds__(256) void k_influence(
    const float* __restrict__ C, const float* __restrict__ A,
    float* __restrict__ out)
{
    const int blk = blockIdx.x;                 // < 16*32*32
    const int u = blk & 31, t = (blk >> 5) & 31, b = blk >> 10;
    const int tid = threadIdx.x;
    __shared__ float sct[64], scu[64];
    if (tid < 64)       sct[tid]      = C[(b * 32 + t) * 64 + tid];
    else if (tid < 128) scu[tid - 64] = C[(b * 32 + u) * 64 + (tid - 64)] *
                                        A[(b * 32 + t) * 32 + u];
    __syncthreads();
    const int wave = tid >> 5, lane = tid & 31;
    const size_t base = (size_t)(b * 32 + t) * 131072 + (size_t)u * 64;
    const float j0 = scu[lane * 2], j1 = scu[lane * 2 + 1];
    for (int i = wave; i < 64; i += 8) {
        float ci = sct[i];
        vf2 val; val.x = ci * j0; val.y = ci * j1;
        vf2* p = (vf2*)(out + base + (size_t)i * 2048 + lane * 2);
        __builtin_nontemporal_store(val, p);
    }
}

// ---------------------------------------------------------------------------
// Host-side launch. Input order: depth-first dict insertion order of
// setup_inputs(): x, in_w, in_b, {la_f,la_r,lb_f,lb_r}x{wih,whh,bih,bhh},
// fa_w, fa_b, fb_w, fb_b, layers[0..1]x12, c1_w, c1_b, c2_w, c2_b  (51 total)
// ---------------------------------------------------------------------------
extern "C" void kernel_launch(void* const* d_in, const int* in_sizes, int n_in,
                              void* d_out, int out_size, void* d_ws, size_t ws_size,
                              hipStream_t stream)
{
    (void)in_sizes; (void)n_in; (void)out_size; (void)ws_size;
    const float* x    = (const float*)d_in[0];
    const float* in_w = (const float*)d_in[1];
    const float* in_b = (const float*)d_in[2];
    const float* wih[4] = {(const float*)d_in[3],  (const float*)d_in[7],
                           (const float*)d_in[11], (const float*)d_in[15]};
    const float* whh[4] = {(const float*)d_in[4],  (const float*)d_in[8],
                           (const float*)d_in[12], (const float*)d_in[16]};
    const float* bih[4] = {(const float*)d_in[5],  (const float*)d_in[9],
                           (const float*)d_in[13], (const float*)d_in[17]};
    const float* bhh[4] = {(const float*)d_in[6],  (const float*)d_in[10],
                           (const float*)d_in[14], (const float*)d_in[18]};
    const float* fa_w = (const float*)d_in[19];
    const float* fa_b = (const float*)d_in[20];
    const float* fb_w = (const float*)d_in[21];
    const float* fb_b = (const float*)d_in[22];
    const float* c1_w = (const float*)d_in[47];
    const float* c1_b = (const float*)d_in[48];
    const float* c2_w = (const float*)d_in[49];
    const float* c2_b = (const float*)d_in[50];

    float* out = (float*)d_out;
    float* o_logits = out + 0;           // 16
    float* o_alpha  = out + 16;          // 512
    float* o_beta   = out + 528;         // 32768
    float* o_C      = out + 33296;       // 32768
    float* o_A      = out + 66064;       // 16384
    float* o_inf    = out + 82448;       // 67108864

    // --- f16 weight region (halves), padded to 128-row multiples ---
    _Float16* h16 = (_Float16*)d_ws;
    _Float16* w16_in   = h16;                       //  8192 (128x64)
    _Float16* w16_wih[4], *w16_whh[4];
    for (int d = 0; d < 4; ++d) w16_wih[d] = h16 + 8192 + d * 65536;
    for (int d = 0; d < 4; ++d) w16_whh[d] = h16 + 270336 + d * 65536;
    _Float16* w16_fb   = h16 + 532480;              // 32768 (128x256 padded)
    _Float16* w16_L[2][4];
    for (int L = 0; L < 2; ++L) {
        _Float16* base = h16 + 565248 + L * 196608;
        w16_L[L][0] = base;                         // in_w  384x128
        w16_L[L][1] = base + 49152;                 // out_w 128x128
        w16_L[L][2] = base + 65536;                 // l1_w  512x128
        w16_L[L][3] = base + 131072;                // l2_w  128x512
    }
    _Float16* w16_c1 = h16 + 958464;                // 16384 (128x128 padded)
    // f32 scratch region starts at half-index 974848 -> float offset 487424
    float* wf   = (float*)d_ws + 487424;
    float* E    = wf;                               // 65536 (reused as xc)
    float* xp[4] = {wf + 65536, wf + 65536 + 262144,
                    wf + 65536 + 2 * 262144, wf + 65536 + 3 * 262144};
    float* outa = wf + 65536 + 4 * 262144;          // 131072
    float* outb = outa + 131072;                    // 131072
    float* qkvb = outb + 131072;                    // 196608
    float* obuf = qkvb + 196608;                    // 65536
    float* tmp  = obuf + 65536;                     // 65536
    float* ffb  = tmp + 65536;                      // 262144
    float* ctx  = ffb + 262144;                     // 2048
    float* hmid = ctx + 2048;                       // 1024

    dim3 blk(256);
    auto cvt = [&](const float* s, _Float16* dst, int n_src, int n_tot) {
        k_cvt<<<dim3((n_tot + 255) / 256), blk, 0, stream>>>(s, dst, n_src, n_tot);
    };

    // weight conversions (f32 -> f16, zero row padding)
    cvt(in_w, w16_in, 8192, 8192);
    for (int d = 0; d < 4; ++d) { cvt(wih[d], w16_wih[d], 65536, 65536);
                                  cvt(whh[d], w16_whh[d], 65536, 65536); }
    cvt(fb_w, w16_fb, 16384, 32768);
    for (int L = 0; L < 2; ++L) {
        const int base = 23 + L * 12;
        cvt((const float*)d_in[base + 0], w16_L[L][0], 49152, 49152);
        cvt((const float*)d_in[base + 2], w16_L[L][1], 16384, 16384);
        cvt((const float*)d_in[base + 4], w16_L[L][2], 65536, 65536);
        cvt((const float*)d_in[base + 6], w16_L[L][3], 65536, 65536);
    }
    cvt(c1_w, w16_c1, 8192, 16384);

    // E = x @ in_w^T + in_b ; then += positional encoding
    k_gemm<<<dim3(32, 1), blk, 0, stream>>>(x, w16_in, in_b, nullptr, E, 512, 128, 64, 0);
    k_add_pe<<<dim3(256), blk, 0, stream>>>(E);

    // LSTM input projections (bias = bih + bhh), then 4 persistent recurrences
    for (int d = 0; d < 4; ++d)
        k_gemm<<<dim3(32, 4), blk, 0, stream>>>(E, w16_wih[d], bih[d], bhh[d], xp[d],
                                                512, 512, 128, 0);
    k_lstm<<<dim3(4), blk, 0, stream>>>(xp[0], xp[1], xp[2], xp[3],
                                        w16_whh[0], w16_whh[1], w16_whh[2], w16_whh[3],
                                        outa, outb);

    // alpha, beta, C
    k_alpha<<<dim3(16), blk, 0, stream>>>(outa, fa_w, fa_b, o_alpha);
    k_gemm<<<dim3(32, 1), blk, 0, stream>>>(outb, w16_fb, fb_b, nullptr, o_beta,
                                            512, 64, 256, 2);
    k_C<<<dim3(128), blk, 0, stream>>>(x, o_alpha, o_beta, o_C);

    // transformer (xc aliases E)
    float* xc = E;
    for (int L = 0; L < 2; ++L) {
        const int base = 23 + L * 12;
        const float* L_in_b  = (const float*)d_in[base + 1];
        const float* L_out_b = (const float*)d_in[base + 3];
        const float* L_l1_b  = (const float*)d_in[base + 5];
        const float* L_l2_b  = (const float*)d_in[base + 7];
        const float* L_n1_g  = (const float*)d_in[base + 8];
        const float* L_n1_b  = (const float*)d_in[base + 9];
        const float* L_n2_g  = (const float*)d_in[base + 10];
        const float* L_n2_b  = (const float*)d_in[base + 11];

        k_gemm<<<dim3(32, 3), blk, 0, stream>>>(xc, w16_L[L][0], L_in_b, nullptr, qkvb,
                                                512, 384, 128, 0);
        k_attn<<<dim3(16), blk, 0, stream>>>(qkvb, obuf, o_A, L);
        k_gemm<<<dim3(32, 1), blk, 0, stream>>>(obuf, w16_L[L][1], L_out_b, nullptr, tmp,
                                                512, 128, 128, 0);
        k_addln<<<dim3(512), dim3(128), 0, stream>>>(xc, tmp, L_n1_g, L_n1_b, xc);
        k_gemm<<<dim3(32, 4), blk, 0, stream>>>(xc, w16_L[L][2], L_l1_b, nullptr, ffb,
                                                512, 512, 128, 1);
        k_gemm<<<dim3(32, 1), blk, 0, stream>>>(ffb, w16_L[L][3], L_l2_b, nullptr, tmp,
                                                512, 128, 512, 0);
        k_addln<<<dim3(512), dim3(128), 0, stream>>>(xc, tmp, L_n2_g, L_n2_b, xc);
    }

    // classifier head
    k_context<<<dim3(16), dim3(128), 0, stream>>>(xc, o_alpha, ctx);
    k_gemm<<<dim3(1, 1), blk, 0, stream>>>(ctx, w16_c1, c1_b, nullptr, hmid,
                                           16, 64, 128, 1);
    k_logits<<<dim3(1), dim3(32), 0, stream>>>(hmid, c2_w, c2_b, o_logits);

    // the 268 MB streaming einsum
    k_influence<<<dim3(16384), blk, 0, stream>>>(o_C, o_A, o_inf);
}
